// MCA_59425167508151
// MI455X (gfx1250) — matrix operand order repaired and checked by reading, not compiled, not run on hardware
//
#include <hip/hip_runtime.h>
#include <hip/hip_bf16.h>

// ---------------------------------------------------------------------------
// Problem constants (B,C,U,V,H,W) = (4,64,5,5,64,64), MID=64
// ---------------------------------------------------------------------------
#define NB   4
#define NC   64
#define NU   5
#define NV   5
#define NH   64
#define NW   64
#define PLANE (NH*NW)          // 4096
#define NPLN  (NU*NV)          // 25
#define SLICE (NPLN*PLANE)     // 102400 floats per (b,c)
#define NBC   (NB*NC)          // 256

// region position counts
#define NP_HW (NH*NW)          // 4096
#define NP_VW (NV*NW)          // 320
#define NP_UH (NU*NH)          // 320
#define NP_UV (NU*NV)          // 25
#define TILES_PER_B 298        // 256 + 20 + 20 + 2

typedef __attribute__((ext_vector_type(16))) _Float16 v16h;
typedef __attribute__((ext_vector_type(8)))  _Float16 h8;
typedef __attribute__((ext_vector_type(8)))  float    v8f;
typedef __attribute__((ext_vector_type(4)))  float    f4;

// Assemble a 16-half WMMA fragment from two 16B chunks 32B apart.
// Layout (A 16x32 f16 / B 32x16 f16 per ISA): lane half l>>4 selects +8 K,
// halves [0..7] = K[kb+8*hi .. +7], halves [8..15] = K[kb+16+8*hi .. +7].
__device__ inline v16h frag16(const _Float16* row_kb, int hi) {
  h8 lo = *(const h8*)(row_kb + 8 * hi);
  h8 hp = *(const h8*)(row_kb + 16 + 8 * hi);
  return __builtin_shufflevector(lo, hp, 0, 1, 2, 3, 4, 5, 6, 7,
                                 8, 9, 10, 11, 12, 13, 14, 15);
}

// ---------------------------------------------------------------------------
// Kernel 0: convert all six 64x64 weight matrices to f16 (row-major kept:
// row = output channel, col = K). 24576 elements total.
// ---------------------------------------------------------------------------
__global__ __launch_bounds__(256) void k_wcvt(
    const float* __restrict__ w1, const float* __restrict__ w2,
    const float* __restrict__ f0, const float* __restrict__ f1,
    const float* __restrict__ f2, const float* __restrict__ f3,
    _Float16* __restrict__ wh) {
  int i = blockIdx.x * 256 + threadIdx.x;   // 0..24575
  const float* src;
  int o;
  if (i < 4096)      { src = w1; o = i; }
  else if (i < 8192) { src = w2; o = i - 4096; }
  else {
    int m = (i - 8192) >> 12;
    o = (i - 8192) & 4095;
    src = (m == 0) ? f0 : (m == 1) ? f1 : (m == 2) ? f2 : f3;
  }
  wh[i] = (_Float16)src[o];
}

// ---------------------------------------------------------------------------
// Kernel 1: fused 4-way axis reduction. One block per (b,c) slice (400KB
// contiguous). Thread t owns w-quad (t&15)*4 and h rows t>>4 + {0,16,32,48}.
// ---------------------------------------------------------------------------
__global__ __launch_bounds__(256) void k_reduce(
    const float* __restrict__ x,
    float* __restrict__ hw_m, float* __restrict__ uh_m,
    float* __restrict__ vw_m, float* __restrict__ uv_m) {
  __shared__ float uv_l[NPLN];
  __shared__ float uh_l[NP_UH];
  __shared__ float vw_l[NP_VW];
  const int t = threadIdx.x;
  const int bc = blockIdx.x;

  for (int i = t; i < 320; i += 256) { uh_l[i] = 0.f; vw_l[i] = 0.f; }
  if (t < NPLN) uv_l[t] = 0.f;
  __syncthreads();

  const float* xs = x + (size_t)bc * SLICE;
  const int w4 = (t & 15) * 4;
  const int hb = t >> 4;

  f4 hwacc[4] = {};
  float uhacc[NU * 4] = {};
  f4 vwacc[NV] = {};

  for (int p = 0; p < NPLN; ++p) {
    const int u = p / NV, v = p % NV;
    const float* pl = xs + p * PLANE;
    float psum = 0.f;
    f4 vac = vwacc[v];
#pragma unroll
    for (int j = 0; j < 4; ++j) {
      f4 x4 = *(const f4*)(pl + (hb + 16 * j) * NW + w4);
      hwacc[j] += x4;
      vac += x4;
      float rs = x4.x + x4.y + x4.z + x4.w;
      uhacc[u * 4 + j] += rs;
      psum += rs;
    }
    vwacc[v] = vac;
    atomicAdd(&uv_l[p], psum);
  }

  float* hwp = hw_m + (size_t)bc * PLANE;
#pragma unroll
  for (int j = 0; j < 4; ++j)
    *(f4*)(hwp + (hb + 16 * j) * NW + w4) = hwacc[j] * (1.f / 25.f);

#pragma unroll
  for (int u = 0; u < NU; ++u)
#pragma unroll
    for (int j = 0; j < 4; ++j)
      atomicAdd(&uh_l[u * NH + hb + 16 * j], uhacc[u * 4 + j]);
#pragma unroll
  for (int v = 0; v < NV; ++v)
#pragma unroll
    for (int k = 0; k < 4; ++k)
      atomicAdd(&vw_l[v * NW + w4 + k], vwacc[v][k]);
  __syncthreads();

  for (int i = t; i < 320; i += 256) {
    uh_m[(size_t)bc * NP_UH + i] = uh_l[i] * (1.f / 320.f);
    vw_m[(size_t)bc * NP_VW + i] = vw_l[i] * (1.f / 320.f);
  }
  if (t < NPLN) uv_m[(size_t)bc * NPLN + t] = uv_l[t] * (1.f / 4096.f);
}

// ---------------------------------------------------------------------------
// Kernel 2: per-position MLP (64 -silu-> 64 -> 64) + region conv, WMMA f16.
// One wave per 16-position tile. 3 layers x 4 N-tiles x 2 K-steps = 24 wmma.
// ---------------------------------------------------------------------------
__global__ __launch_bounds__(32) void k_mlp(
    const float* __restrict__ hw_m, const float* __restrict__ uh_m,
    const float* __restrict__ vw_m, const float* __restrict__ uv_m,
    float* __restrict__ chw, float* __restrict__ cuh,
    float* __restrict__ cvw, float* __restrict__ cuv,
    const _Float16* __restrict__ wh,
    const float* __restrict__ b1, const float* __restrict__ b2,
    const float* __restrict__ fb0, const float* __restrict__ fb1,
    const float* __restrict__ fb2, const float* __restrict__ fb3) {
  __shared__ _Float16 Xl[16 * 72];   // pitch 72 halves: 16B-aligned rows, spread banks
  __shared__ _Float16 Yl[16 * 72];

  const int l = threadIdx.x;
  const int blk = blockIdx.x;
  const int b = blk / TILES_PER_B;
  const int t = blk % TILES_PER_B;

  const float* src; float* dst; const _Float16* fw; const float* fb;
  int Npos, pbase;
  if (t < 256)      { src = hw_m; dst = chw; fw = wh + 8192 + 0 * 4096; fb = fb0; Npos = NP_HW; pbase = t * 16; }
  else if (t < 276) { src = vw_m; dst = cvw; fw = wh + 8192 + 3 * 4096; fb = fb3; Npos = NP_VW; pbase = (t - 256) * 16; }
  else if (t < 296) { src = uh_m; dst = cuh; fw = wh + 8192 + 2 * 4096; fb = fb2; Npos = NP_UH; pbase = (t - 276) * 16; }
  else              { src = uv_m; dst = cuv; fw = wh + 8192 + 1 * 4096; fb = fb1; Npos = NP_UV; pbase = (t - 296) * 16; }

  // Stage X tile [16 pos][64 ch] into LDS as f16. 1024 values / 32 lanes.
  const float* sb = src + (size_t)b * NC * Npos;
  for (int j = 0; j < 32; ++j) {
    int idx = j * 32 + l;
    int c = idx >> 4, po = idx & 15;
    int p = pbase + po;
    p = (p < Npos) ? p : (Npos - 1);          // clamp (uv tile padding)
    Xl[po * 72 + c] = (_Float16)sb[(size_t)c * Npos + p];
  }
  __syncthreads();

  const int n = l & 15;     // A-row (M) AND D-column (N) for this lane
  const int hi = l >> 4;
  const _Float16* w1h = wh;
  const _Float16* w2h = wh + 4096;

  // ---- layer 1: Y = silu(X * W1^T + b1) ----
  for (int nt = 0; nt < 4; ++nt) {
    v8f acc = {};
#pragma unroll
    for (int kb = 0; kb < 64; kb += 32) {
      v16h A = frag16(Xl + n * 72 + kb, hi);
      v16h B = frag16(w1h + (nt * 16 + n) * 64 + kb, hi);
      acc = __builtin_amdgcn_wmma_f32_16x16x32_f16(false, A, false, B,
                                                   (short)0, acc, false, false);
    }
    float bias = b1[nt * 16 + n];
#pragma unroll
    for (int g = 0; g < 8; ++g) {
      float y = acc[g] + bias;
      // silu via fast v_rcp_f32 (result is truncated to f16 anyway)
      y = y * __builtin_amdgcn_rcpf(1.f + __expf(-y));
      Yl[(8 * hi + g) * 72 + nt * 16 + n] = (_Float16)y;
    }
  }
  __syncthreads();

  // ---- layer 2: X = Y * W2^T + b2 ----
  for (int nt = 0; nt < 4; ++nt) {
    v8f acc = {};
#pragma unroll
    for (int kb = 0; kb < 64; kb += 32) {
      v16h A = frag16(Yl + n * 72 + kb, hi);
      v16h B = frag16(w2h + (nt * 16 + n) * 64 + kb, hi);
      acc = __builtin_amdgcn_wmma_f32_16x16x32_f16(false, A, false, B,
                                                   (short)0, acc, false, false);
    }
    float bias = b2[nt * 16 + n];
#pragma unroll
    for (int g = 0; g < 8; ++g)
      Xl[(8 * hi + g) * 72 + nt * 16 + n] = (_Float16)(acc[g] + bias);
  }
  __syncthreads();

  // ---- layer 3: coef = X * fw^T + fb -> global region buffer ----
  for (int nt = 0; nt < 4; ++nt) {
    v8f acc = {};
#pragma unroll
    for (int kb = 0; kb < 64; kb += 32) {
      v16h A = frag16(Xl + n * 72 + kb, hi);
      v16h B = frag16(fw + (nt * 16 + n) * 64 + kb, hi);
      acc = __builtin_amdgcn_wmma_f32_16x16x32_f16(false, A, false, B,
                                                   (short)0, acc, false, false);
    }
    float bias = fb[nt * 16 + n];
    size_t ob = ((size_t)b * NC + nt * 16 + n) * (size_t)Npos;
    if (Npos == NP_UV) {                       // masked scalar stores (padding)
#pragma unroll
      for (int g = 0; g < 8; ++g) {
        int p = pbase + 8 * hi + g;
        if (p < NP_UV) dst[ob + p] = acc[g] + bias;
      }
    } else {                                   // contiguous positions: b128 x2
      f4 lo = {acc[0] + bias, acc[1] + bias, acc[2] + bias, acc[3] + bias};
      f4 hp = {acc[4] + bias, acc[5] + bias, acc[6] + bias, acc[7] + bias};
      *(f4*)(dst + ob + pbase + 8 * hi)     = lo;
      *(f4*)(dst + ob + pbase + 8 * hi + 4) = hp;
    }
  }
}

// ---------------------------------------------------------------------------
// Kernel 3: out = x * (chw[h,w] + cuv[u,v] + cuh[u,h] + cvw[v,w]).
// One block per (b,c); all coefficients for the slice staged in LDS (19KB).
// ---------------------------------------------------------------------------
__global__ __launch_bounds__(256) void k_comb(
    const float* __restrict__ x,
    const float* __restrict__ chw, const float* __restrict__ cuh,
    const float* __restrict__ cvw, const float* __restrict__ cuv,
    float* __restrict__ out) {
  __shared__ float chw_l[PLANE];
  __shared__ float cuh_l[NP_UH];
  __shared__ float cvw_l[NP_VW];
  __shared__ float cuv_l[NPLN];
  const int t = threadIdx.x;
  const int bc = blockIdx.x;

  for (int i = t; i < PLANE / 4; i += 256)
    ((f4*)chw_l)[i] = ((const f4*)(chw + (size_t)bc * PLANE))[i];
  for (int i = t; i < 320; i += 256) {
    cuh_l[i] = cuh[(size_t)bc * NP_UH + i];
    cvw_l[i] = cvw[(size_t)bc * NP_VW + i];
  }
  if (t < NPLN) cuv_l[t] = cuv[(size_t)bc * NPLN + t];
  __syncthreads();

  const float* xs = x + (size_t)bc * SLICE;
  float* os = out + (size_t)bc * SLICE;
  const int w4 = (t & 15) * 4;
  const int hb = t >> 4;

  for (int p = 0; p < NPLN; ++p) {
    const int u = p / NV, v = p % NV;
    const float cs = cuv_l[p];
    const f4 vw4 = *(const f4*)(cvw_l + v * NW + w4);
#pragma unroll
    for (int j = 0; j < 4; ++j) {
      const int h = hb + 16 * j;
      const int idx = h * NW + w4;
      f4 x4 = *(const f4*)(xs + p * PLANE + idx);
      f4 coef = *(const f4*)(chw_l + idx) + vw4 + (cs + cuh_l[u * NH + h]);
      *(f4*)(os + p * PLANE + idx) = x4 * coef;
    }
  }
}

// ---------------------------------------------------------------------------
// Workspace layout (floats, then f16 weights)
// ---------------------------------------------------------------------------
#define WS_HW   0
#define WS_UH   (WS_HW + NBC * NP_HW)       // 1048576
#define WS_VW   (WS_UH + NBC * NP_UH)       // +81920
#define WS_UV   (WS_VW + NBC * NP_VW)       // +81920
#define WS_CHW  (WS_UV + NBC * NP_UV)       // +6400
#define WS_CUH  (WS_CHW + NBC * NP_HW)
#define WS_CVW  (WS_CUH + NBC * NP_UH)
#define WS_CUV  (WS_CVW + NBC * NP_VW)
#define WS_FEND (WS_CUV + NBC * NP_UV)      // 2437632 floats
#define WS_BYTES ((size_t)WS_FEND * 4 + (size_t)24576 * 2)

extern "C" void kernel_launch(void* const* d_in, const int* in_sizes, int n_in,
                              void* d_out, int out_size, void* d_ws, size_t ws_size,
                              hipStream_t stream) {
  (void)in_sizes; (void)n_in; (void)out_size;
  if (ws_size < WS_BYTES) return;

  const float* x   = (const float*)d_in[0];
  const float* w1  = (const float*)d_in[1];
  const float* b1  = (const float*)d_in[2];
  const float* w2  = (const float*)d_in[3];
  const float* b2  = (const float*)d_in[4];
  const float* fw0 = (const float*)d_in[5];
  const float* fb0 = (const float*)d_in[6];
  const float* fw1 = (const float*)d_in[7];
  const float* fb1 = (const float*)d_in[8];
  const float* fw2 = (const float*)d_in[9];
  const float* fb2 = (const float*)d_in[10];
  const float* fw3 = (const float*)d_in[11];
  const float* fb3 = (const float*)d_in[12];
  float* out = (float*)d_out;

  float* wsf = (float*)d_ws;
  float* hw_m = wsf + WS_HW;
  float* uh_m = wsf + WS_UH;
  float* vw_m = wsf + WS_VW;
  float* uv_m = wsf + WS_UV;
  float* chw  = wsf + WS_CHW;
  float* cuh  = wsf + WS_CUH;
  float* cvw  = wsf + WS_CVW;
  float* cuv  = wsf + WS_CUV;
  _Float16* wh = (_Float16*)(wsf + WS_FEND);

  k_wcvt<<<96, 256, 0, stream>>>(w1, w2, fw0, fw1, fw2, fw3, wh);
  k_reduce<<<NBC, 256, 0, stream>>>(x, hw_m, uh_m, vw_m, uv_m);
  k_mlp<<<NB * TILES_PER_B, 32, 0, stream>>>(hw_m, uh_m, vw_m, uv_m,
                                             chw, cuh, cvw, cuv, wh,
                                             b1, b2, fb0, fb1, fb2, fb3);
  k_comb<<<NBC, 256, 0, stream>>>(x, chw, cuh, cvw, cuv, out);
}
